// IPMPEncoder_49959059587655
// MI455X (gfx1250) — matrix operand order repaired and checked by reading, not compiled, not run on hardware
//
#include <hip/hip_runtime.h>

// ---------------------------------------------------------------------------
// Problem constants (from the reference)
// ---------------------------------------------------------------------------
#define NN   4096
#define KK   30
#define EE   (NN * KK)          // 122880 edges
#define CS   256
#define CZ   128
#define CH   256
#define PP   4
#define LL   4
#define ZIN  416                // CZ_IN
#define ZINP 448                // padded to mult of 32
#define MIN_ 656                // MSG_IN
#define MINP 672                // padded
#define CATE 640                // 2*CS + CZ (already mult of 32)

// ---------------------------------------------------------------------------
// WMMA plumbing (CDNA5, wave32, v_wmma_f32_16x16x32_bf16)
// ---------------------------------------------------------------------------
typedef __attribute__((ext_vector_type(16))) __bf16 v16bf;
typedef __attribute__((ext_vector_type(8)))  float  v8f;

__device__ __forceinline__ __bf16 f2bf(float f) {
    unsigned u = __float_as_uint(f);
    unsigned r = (u + 0x7FFFu + ((u >> 16) & 1u)) >> 16;   // RNE
    unsigned short h = (unsigned short)r;
    __bf16 o;
    __builtin_memcpy(&o, &h, 2);
    return o;
}

__device__ __forceinline__ v8f v8f_zero() {
    v8f z = {0.f, 0.f, 0.f, 0.f, 0.f, 0.f, 0.f, 0.f};
    return z;
}

// A fragment (16x32 bf16), row-major LDS tile with element stride `stride`.
// CDNA5 layout: lane m = lane&15, half = lane>>4;
// VGPR r in 0..3 -> K = 2r + 8*half ; r in 4..7 -> +16.
__device__ __forceinline__ v16bf ld_afrag(const __bf16* A, int stride, int lane) {
    int m = lane & 15, half = lane >> 4;
    v16bf a;
#pragma unroll
    for (int r = 0; r < 8; ++r) {
        int kb = 2 * (r & 3) + half * 8 + (r >> 2) * 16;
        a[2 * r]     = A[m * stride + kb];
        a[2 * r + 1] = A[m * stride + kb + 1];
    }
    return a;
}

// B fragment (32x16 bf16) from a W^T tile stored [n][k], k-stride 32.
// VGPR r in 0..3 -> K = 2r + 8*half ; r in 4..7 -> +16 ; n = lane&15.
__device__ __forceinline__ v16bf ld_bfrag(const __bf16* Wt, int lane) {
    int n = lane & 15, kh = lane >> 4;
    v16bf b;
#pragma unroll
    for (int r = 0; r < 8; ++r) {
        int k = 2 * (r & 3) + kh * 8 + (r >> 2) * 16;
        b[2 * r]     = Wt[n * 32 + k];
        b[2 * r + 1] = Wt[n * 32 + k + 1];
    }
    return b;
}

__device__ __forceinline__ v8f wmma_bf16(v16bf a, v16bf b, v8f c) {
    return __builtin_amdgcn_wmma_f32_16x16x32_bf16(false, a, false, b,
                                                   (short)0, c, false, false);
}

// ---------------------------------------------------------------------------
// small helpers
// ---------------------------------------------------------------------------
__device__ __forceinline__ void unit3(float* v) {
    float n = sqrtf(v[0] * v[0] + v[1] * v[1] + v[2] * v[2]) + 1e-8f;
    v[0] /= n; v[1] /= n; v[2] /= n;
}
__device__ __forceinline__ void cross3(const float* a, const float* b, float* o) {
    o[0] = a[1] * b[2] - a[2] * b[1];
    o[1] = a[2] * b[0] - a[0] * b[2];
    o[2] = a[0] * b[1] - a[1] * b[0];
}

// ---------------------------------------------------------------------------
// weight f32 -> bf16 (K padded with zeros to kp)
// ---------------------------------------------------------------------------
__global__ void cvt_w_kernel(const float* __restrict__ src, __bf16* __restrict__ dst,
                             int rows, int kin, int kp) {
    int i = blockIdx.x * 256 + threadIdx.x;
    if (i >= rows * kp) return;
    int r = i / kp, c = i % kp;
    dst[i] = (c < kin) ? f2bf(src[(size_t)r * kin + c]) : f2bf(0.f);
}

__global__ void fill_zero_kernel(float* __restrict__ p, int n) {
    int i = blockIdx.x * 256 + threadIdx.x;
    if (i < n) p[i] = 0.f;
}

// ---------------------------------------------------------------------------
// prep: bb5 (N x 5 x 3) and dihedral node features (N x 6)
// ---------------------------------------------------------------------------
__global__ void prep_kernel(const float* __restrict__ bb, float* __restrict__ bb5,
                            float* __restrict__ nfeat) {
    int n = blockIdx.x * 128 + threadIdx.x;
    if (n >= NN) return;
    const float* nb = bb + (size_t)n * 12;
    float Na[3], CA[3], Cc[3], bv[3], cd[3], av[3];
#pragma unroll
    for (int j = 0; j < 3; ++j) { Na[j] = nb[j]; CA[j] = nb[3 + j]; Cc[j] = nb[6 + j]; }
#pragma unroll
    for (int j = 0; j < 3; ++j) { bv[j] = CA[j] - Na[j]; cd[j] = Cc[j] - CA[j]; }
    cross3(bv, cd, av);
#pragma unroll
    for (int t = 0; t < 4; ++t)
#pragma unroll
        for (int j = 0; j < 3; ++j)
            bb5[(size_t)n * 15 + t * 3 + j] = bb[(size_t)n * 12 + t * 3 + j];
#pragma unroll
    for (int j = 0; j < 3; ++j)
        bb5[(size_t)n * 15 + 12 + j] =
            -0.58273431f * av[j] + 0.56802827f * bv[j] - 0.54067466f * cd[j] + CA[j];

    // dihedrals
    for (int jj = 0; jj < 3; ++jj) {
        int u = 3 * n + jj - 1;
        float cD = 1.f, sD = 0.f;
        if (u >= 0 && u <= 3 * NN - 4) {
            float X[4][3];
#pragma unroll
            for (int tt = 0; tt < 4; ++tt) {
                int t = u + tt, row = t / 3, at = t % 3;
#pragma unroll
                for (int j = 0; j < 3; ++j)
                    X[tt][j] = bb[(size_t)row * 12 + at * 3 + j];
            }
            float u2[3], u1[3], u0[3], n2[3], n1[3];
#pragma unroll
            for (int j = 0; j < 3; ++j) {
                u2[j] = X[1][j] - X[0][j];
                u1[j] = X[2][j] - X[1][j];
                u0[j] = X[3][j] - X[2][j];
            }
            unit3(u2); unit3(u1); unit3(u0);
            cross3(u2, u1, n2); unit3(n2);
            cross3(u1, u0, n1); unit3(n1);
            float cosD = n2[0] * n1[0] + n2[1] * n1[1] + n2[2] * n1[2];
            cosD = fminf(fmaxf(cosD, -1.f + 1e-7f), 1.f - 1e-7f);
            float sg = u2[0] * n1[0] + u2[1] * n1[1] + u2[2] * n1[2];
            float sgn = (sg > 0.f) ? 1.f : ((sg < 0.f) ? -1.f : 0.f);
            float D = sgn * acosf(cosD);
            cD = cosf(D); sD = sinf(D);
        }
        nfeat[(size_t)n * 6 + jj] = cD;
        nfeat[(size_t)n * 6 + 3 + jj] = sD;
    }
}

// ---------------------------------------------------------------------------
// KNN: one block per query node; d2 in LDS, K rounds of argmin
// ---------------------------------------------------------------------------
__global__ void knn_kernel(const float* __restrict__ xca,
                           const unsigned char* __restrict__ mask,
                           const int* __restrict__ batch,
                           int* __restrict__ eidx) {
    __shared__ float d2[NN];
    __shared__ float rmin[128];
    __shared__ int   rid[128];
    int n = blockIdx.x, tid = threadIdx.x;
    float qx = xca[(size_t)n * 3], qy = xca[(size_t)n * 3 + 1], qz = xca[(size_t)n * 3 + 2];
    if (!mask[n]) { qx = 1e6f; qy = 1e6f; qz = 1e6f; }
    int qb = batch[n];
    for (int i = tid; i < NN; i += 128) {
        float ix = xca[(size_t)i * 3], iy = xca[(size_t)i * 3 + 1], iz = xca[(size_t)i * 3 + 2];
        if (!mask[i]) { ix = 1e6f; iy = 1e6f; iz = 1e6f; }
        float dx = qx - ix, dy = qy - iy, dz = qz - iz;
        float d = dx * dx + dy * dy + dz * dz;
        if (batch[i] != qb || i == n) d = 1e12f;
        d2[i] = d;
    }
    __syncthreads();
    for (int j = 0; j < KK; ++j) {
        float best = 3e38f; int bid = 0;
        for (int i = tid; i < NN; i += 128) {
            float d = d2[i];
            if (d < best) { best = d; bid = i; }
        }
        rmin[tid] = best; rid[tid] = bid;
        __syncthreads();
        for (int o = 64; o > 0; o >>= 1) {
            if (tid < o) {
                float a = rmin[tid], b = rmin[tid + o];
                if (b < a || (b == a && rid[tid + o] < rid[tid])) {
                    rmin[tid] = b; rid[tid] = rid[tid + o];
                }
            }
            __syncthreads();
        }
        if (tid == 0) {
            eidx[(size_t)n * KK + j] = rid[0];
            d2[rid[0]] = 3e38f;
        }
        __syncthreads();
    }
}

// ---------------------------------------------------------------------------
// rigid frames R (N x 9), t (N x 3)
// ---------------------------------------------------------------------------
__global__ void rigid_kernel(const float* __restrict__ rig,
                             float* __restrict__ Rm, float* __restrict__ tv) {
    int n = blockIdx.x * 128 + threadIdx.x;
    if (n >= NN) return;
    const float* q = rig + (size_t)n * 7;
    float nrm = sqrtf(q[0] * q[0] + q[1] * q[1] + q[2] * q[2] + q[3] * q[3]);
    float w = q[0] / nrm, x = q[1] / nrm, y = q[2] / nrm, z = q[3] / nrm;
    float* R = Rm + (size_t)n * 9;
    R[0] = 1.f - 2.f * (y * y + z * z); R[1] = 2.f * (x * y - w * z); R[2] = 2.f * (x * z + w * y);
    R[3] = 2.f * (x * y + w * z); R[4] = 1.f - 2.f * (x * x + z * z); R[5] = 2.f * (y * z - w * x);
    R[6] = 2.f * (x * z - w * y); R[7] = 2.f * (y * z + w * x); R[8] = 1.f - 2.f * (x * x + y * y);
    tv[(size_t)n * 3 + 0] = q[4];
    tv[(size_t)n * 3 + 1] = q[5];
    tv[(size_t)n * 3 + 2] = q[6];
}

// ---------------------------------------------------------------------------
// pts = s @ qw^T + qb ; g = R*pts + t     (tiny -> scalar VALU)
// ---------------------------------------------------------------------------
__global__ void pts_kernel(const float* __restrict__ s, const float* __restrict__ qw,
                           const float* __restrict__ qb, const float* __restrict__ Rm,
                           const float* __restrict__ tv, float* __restrict__ g) {
    int t = blockIdx.x * 128 + threadIdx.x;
    if (t >= NN * PP) return;
    int n = t / PP, p = t % PP;
    const float* sr = s + (size_t)n * CS;
    float pt[3];
#pragma unroll
    for (int j = 0; j < 3; ++j) {
        float acc = qb[p * 3 + j];
        const float* wr = qw + (size_t)(p * 3 + j) * CS;
        for (int c = 0; c < CS; ++c) acc += wr[c] * sr[c];
        pt[j] = acc;
    }
    const float* R = Rm + (size_t)n * 9;
#pragma unroll
    for (int i = 0; i < 3; ++i)
        g[(size_t)n * 12 + p * 3 + i] =
            R[i * 3 + 0] * pt[0] + R[i * 3 + 1] * pt[1] + R[i * 3 + 2] * pt[2] +
            tv[(size_t)n * 3 + i];
}

// ---------------------------------------------------------------------------
// generic node-level GEMM: C[MxNc] = act(A[MxK] @ W^T + bias), W bf16 [Nc x ldw]
// block 128 thr (4 waves); tile 32x64; each wave 16x32 -> 2 accumulators
// ---------------------------------------------------------------------------
__global__ void gemm_kernel(const float* __restrict__ A, int lda, int M, int Kdim,
                            const __bf16* __restrict__ W, int ldw,
                            const float* __restrict__ bias,
                            float* __restrict__ C, int ldc, int Ncols, int relu) {
    __shared__ __attribute__((aligned(16))) __bf16 sA[32 * 32];
    __shared__ __attribute__((aligned(16))) __bf16 sW[64 * 32];
    const int tid = threadIdx.x, lane = tid & 31, wv = tid >> 5;
    const int m0 = blockIdx.x * 32, n0 = blockIdx.y * 64;
    const int rt = (wv >> 1) * 16, ct = (wv & 1) * 32;
    v8f acc0 = v8f_zero(), acc1 = v8f_zero();
    const int Kp = (Kdim + 31) & ~31;
    for (int k0 = 0; k0 < Kp; k0 += 32) {
        __syncthreads();
        for (int i = tid; i < 32 * 32; i += 128) {
            int r = i >> 5, c = i & 31;
            int gr = m0 + r, gk = k0 + c;
            float v = (gr < M && gk < Kdim) ? A[(size_t)gr * lda + gk] : 0.f;
            sA[i] = f2bf(v);
        }
        for (int i = tid; i < 64 * 32; i += 128) {
            int r = i >> 5, c = i & 31;
            int gn = n0 + r;
            sW[i] = (gn < Ncols) ? W[(size_t)gn * ldw + k0 + c] : f2bf(0.f);
        }
        __syncthreads();
        if (k0 + 32 < Kp)
            __builtin_prefetch(&W[(size_t)(n0 + (tid & 63)) * ldw + k0 + 32]);
        v16bf af = ld_afrag(sA + rt * 32, 32, lane);
        v16bf b0 = ld_bfrag(sW + (ct + 0) * 32, lane);
        v16bf b1 = ld_bfrag(sW + (ct + 16) * 32, lane);
        acc0 = wmma_bf16(af, b0, acc0);
        acc1 = wmma_bf16(af, b1, acc1);
    }
    int rbase = m0 + rt + ((lane >> 4) << 3);
    int c0 = n0 + ct + (lane & 15);
#pragma unroll
    for (int v = 0; v < 8; ++v) {
        int row = rbase + v;
        if (row >= M) continue;
        if (c0 < Ncols) {
            float x0 = acc0[v] + (bias ? bias[c0] : 0.f);
            if (relu) x0 = fmaxf(x0, 0.f);
            C[(size_t)row * ldc + c0] = x0;
        }
        if (c0 + 16 < Ncols) {
            float x1 = acc1[v] + (bias ? bias[c0 + 16] : 0.f);
            if (relu) x1 = fmaxf(x1, 0.f);
            C[(size_t)row * ldc + c0 + 16] = x1;
        }
    }
}

// ---------------------------------------------------------------------------
// row LayerNorm (node level): out = LN(x (+res)) * mask ; one block per row
// ---------------------------------------------------------------------------
__global__ void ln_rows_kernel(const float* __restrict__ x, const float* __restrict__ res,
                               float* __restrict__ out, const float* __restrict__ gam,
                               const float* __restrict__ bet,
                               const unsigned char* __restrict__ mask, int C) {
    __shared__ float red[128];
    int row = blockIdx.x, tid = threadIdx.x;
    const float* xr = x + (size_t)row * C;
    const float* rr = res ? res + (size_t)row * C : nullptr;
    float s0 = 0.f;
    for (int c = tid; c < C; c += 128) s0 += xr[c] + (rr ? rr[c] : 0.f);
    red[tid] = s0; __syncthreads();
    for (int o = 64; o > 0; o >>= 1) { if (tid < o) red[tid] += red[tid + o]; __syncthreads(); }
    float mean = red[0] / (float)C;
    __syncthreads();
    float s1 = 0.f;
    for (int c = tid; c < C; c += 128) {
        float v = xr[c] + (rr ? rr[c] : 0.f) - mean;
        s1 += v * v;
    }
    red[tid] = s1; __syncthreads();
    for (int o = 64; o > 0; o >>= 1) { if (tid < o) red[tid] += red[tid + o]; __syncthreads(); }
    float rstd = rsqrtf(red[0] / (float)C + 1e-5f);
    float mk = mask ? (mask[row] ? 1.f : 0.f) : 1.f;
    for (int c = tid; c < C; c += 128) {
        float v = xr[c] + (rr ? rr[c] : 0.f);
        out[(size_t)row * C + c] = ((v - mean) * rstd * gam[c] + bet[c]) * mk;
    }
}

__global__ void cat_kernel(const float* __restrict__ s, const float* __restrict__ agg,
                           float* __restrict__ out) {
    int i = blockIdx.x * 256 + threadIdx.x;
    if (i >= NN * 512) return;
    int r = i >> 9, c = i & 511;
    out[i] = (c < 256) ? s[(size_t)r * 256 + c] : agg[(size_t)r * 256 + (c - 256)];
}

// ---------------------------------------------------------------------------
// FUSED edge-feature + 3-layer edge MLP + LN -> z     (32 edges per block)
// LDS overlay (61440B): [0:Az 32x448 bf16 | later H2 32x256]
//                       [28672: dbuf 32x25 f32 / sWt 256x32 bf16]
//                       [45056: H1 32x256 bf16 | later outF 32x128 f32]
// ---------------------------------------------------------------------------
__global__ void edge_mlp_kernel(const float* __restrict__ bb5,
                                const int* __restrict__ eidx,
                                const __bf16* __restrict__ w0, const float* __restrict__ b0,
                                const __bf16* __restrict__ w1, const float* __restrict__ b1,
                                const __bf16* __restrict__ w2, const float* __restrict__ b2,
                                const float* __restrict__ elg, const float* __restrict__ elb,
                                float* __restrict__ z) {
    __shared__ __attribute__((aligned(16))) char smem[61440];
    __shared__ int sSrc[32], sDst[32];
    __shared__ float lmean[32], lrstd[32];
    __bf16* Az   = (__bf16*)smem;
    __bf16* sWt  = (__bf16*)(smem + 28672);
    __bf16* H1   = (__bf16*)(smem + 45056);
    __bf16* H2   = (__bf16*)smem;
    float*  dbuf = (float*)(smem + 28672);
    float*  outF = (float*)(smem + 45056);
    int tid = threadIdx.x, lane = tid & 31, wv = tid >> 5;
    int e0 = blockIdx.x * 32;
    if (tid < 32) { int e = e0 + tid; sDst[tid] = e / KK; sSrc[tid] = eidx[e]; }
    __syncthreads();
    // pairwise atom distances (5x5 per edge)
    for (int i = tid; i < 32 * 25; i += 128) {
        int r = i / 25, pr = i % 25, aa = pr / 5, bbx = pr % 5;
        const float* ps = bb5 + (size_t)sSrc[r] * 15 + aa * 3;
        const float* pd = bb5 + (size_t)sDst[r] * 15 + bbx * 3;
        float dx = ps[0] - pd[0] + 1e-8f;
        float dy = ps[1] - pd[1] + 1e-8f;
        float dz = ps[2] - pd[2] + 1e-8f;
        dbuf[i] = sqrtf(dx * dx + dy * dy + dz * dz);
    }
    __syncthreads();
    // RBF + positional encoding -> Az (bf16)
    for (int i = tid; i < 32 * ZINP; i += 128) {
        int r = i / ZINP, c = i % ZINP;
        float v = 0.f;
        if (c < 400) {
            int pr = c >> 4, q = c & 15;
            float mu = 2.f + (20.f / 15.f) * (float)q;
            float tt = (dbuf[r * 25 + pr] - mu) * 0.8f;     // /1.25
            v = __expf(-tt * tt);
        } else if (c < 416) {
            int f = (c - 400) & 7;
            float dd = (float)(sDst[r] - sSrc[r]);
            float fr = __expf(-(float)(2 * f) * 0.5756462732485115f);  // ln(1e4)/16
            float ang = dd * fr;
            v = (c < 408) ? __cosf(ang) : __sinf(ang);
        }
        Az[i] = f2bf(v);
    }
    // ---- stage 1: (32x448) @ w0^T -> H1 (32x256), relu
    {
        v8f acc[8];
#pragma unroll
        for (int j = 0; j < 8; ++j) acc[j] = v8f_zero();
        int rt = (wv >> 1) * 16, ct = (wv & 1) * 128;
        for (int k0 = 0; k0 < ZINP; k0 += 32) {
            __syncthreads();
            for (int i = tid; i < 256 * 32; i += 128) {
                int r = i >> 5, c = i & 31;
                sWt[i] = w0[(size_t)r * ZINP + k0 + c];
            }
            __syncthreads();
            v16bf af = ld_afrag(Az + rt * ZINP + k0, ZINP, lane);
#pragma unroll
            for (int j = 0; j < 8; ++j)
                acc[j] = wmma_bf16(af, ld_bfrag(sWt + (ct + j * 16) * 32, lane), acc[j]);
        }
        int rbase = rt + ((lane >> 4) << 3), cb = ct + (lane & 15);
#pragma unroll
        for (int j = 0; j < 8; ++j)
#pragma unroll
            for (int v = 0; v < 8; ++v) {
                int row = rbase + v, col = cb + j * 16;
                H1[row * 256 + col] = f2bf(fmaxf(acc[j][v] + b0[col], 0.f));
            }
    }
    // ---- stage 2: H1 @ w1^T -> H2 (32x256), relu
    {
        v8f acc[8];
#pragma unroll
        for (int j = 0; j < 8; ++j) acc[j] = v8f_zero();
        int rt = (wv >> 1) * 16, ct = (wv & 1) * 128;
        for (int k0 = 0; k0 < 256; k0 += 32) {
            __syncthreads();
            for (int i = tid; i < 256 * 32; i += 128) {
                int r = i >> 5, c = i & 31;
                sWt[i] = w1[(size_t)r * 256 + k0 + c];
            }
            __syncthreads();
            v16bf af = ld_afrag(H1 + rt * 256 + k0, 256, lane);
#pragma unroll
            for (int j = 0; j < 8; ++j)
                acc[j] = wmma_bf16(af, ld_bfrag(sWt + (ct + j * 16) * 32, lane), acc[j]);
        }
        __syncthreads();
        int rbase = rt + ((lane >> 4) << 3), cb = ct + (lane & 15);
#pragma unroll
        for (int j = 0; j < 8; ++j)
#pragma unroll
            for (int v = 0; v < 8; ++v) {
                int row = rbase + v, col = cb + j * 16;
                H2[row * 256 + col] = f2bf(fmaxf(acc[j][v] + b1[col], 0.f));
            }
    }
    // ---- stage 3: H2 @ w2^T -> outF (32x128)
    {
        v8f acc[4];
#pragma unroll
        for (int j = 0; j < 4; ++j) acc[j] = v8f_zero();
        int rt = (wv >> 1) * 16, ct = (wv & 1) * 64;
        for (int k0 = 0; k0 < 256; k0 += 32) {
            __syncthreads();
            for (int i = tid; i < 128 * 32; i += 128) {
                int r = i >> 5, c = i & 31;
                sWt[i] = w2[(size_t)r * 256 + k0 + c];
            }
            __syncthreads();
            v16bf af = ld_afrag(H2 + rt * 256 + k0, 256, lane);
#pragma unroll
            for (int j = 0; j < 4; ++j)
                acc[j] = wmma_bf16(af, ld_bfrag(sWt + (ct + j * 16) * 32, lane), acc[j]);
        }
        __syncthreads();
        int rbase = rt + ((lane >> 4) << 3), cb = ct + (lane & 15);
#pragma unroll
        for (int j = 0; j < 4; ++j)
#pragma unroll
            for (int v = 0; v < 8; ++v) {
                int row = rbase + v, col = cb + j * 16;
                outF[row * 128 + col] = acc[j][v] + b2[col];
            }
    }
    __syncthreads();
    // LayerNorm -> z
    if (tid < 32) {
        float m = 0.f;
        for (int c = 0; c < 128; ++c) m += outF[tid * 128 + c];
        m *= (1.f / 128.f);
        float vv = 0.f;
        for (int c = 0; c < 128; ++c) {
            float d = outF[tid * 128 + c] - m;
            vv += d * d;
        }
        lmean[tid] = m;
        lrstd[tid] = rsqrtf(vv * (1.f / 128.f) + 1e-5f);
    }
    __syncthreads();
    for (int i = tid; i < 32 * 128; i += 128) {
        int r = i >> 7, c = i & 127;
        z[(size_t)(e0 + r) * CZ + c] = (outF[i] - lmean[r]) * lrstd[r] * elg[c] + elb[c];
    }
}

// ---------------------------------------------------------------------------
// FUSED message kernel: mi gather -> relu GEMM (672->256) -> GEMM (256->256)
// -> atomic mean into agg.   16 edges per block, 4 waves x (16x64) tiles.
// ---------------------------------------------------------------------------
__global__ void msg_kernel(const float* __restrict__ s, const float* __restrict__ z,
                           const int* __restrict__ eidx, const float* __restrict__ g,
                           const float* __restrict__ Rm, const float* __restrict__ tv,
                           const __bf16* __restrict__ w1, const float* __restrict__ b1,
                           const __bf16* __restrict__ w2, const float* __restrict__ b2,
                           float* __restrict__ agg) {
    __shared__ __attribute__((aligned(16))) char smem[47104];
    __shared__ int sSrc[16], sDst[16];
    __bf16* Ami  = (__bf16*)smem;              // 16 x 672
    __bf16* sWt  = (__bf16*)(smem + 21504);    // 256 x 32
    __bf16* H    = (__bf16*)(smem + 37888);    // 16 x 256
    float*  relT = (float*)(smem + 46080);     // 16 x 16 (rel[12] | dist[4])
    int tid = threadIdx.x, lane = tid & 31, wv = tid >> 5;
    int e0 = blockIdx.x * 16;
    if (tid < 16) { int e = e0 + tid; sDst[tid] = e / KK; sSrc[tid] = eidx[e]; }
    __syncthreads();
    if (tid < 64) {
        int r = tid >> 2, p = tid & 3;
        int src = sSrc[r], dst = sDst[r];
        float v0 = g[(size_t)src * 12 + p * 3 + 0] - tv[(size_t)dst * 3 + 0];
        float v1 = g[(size_t)src * 12 + p * 3 + 1] - tv[(size_t)dst * 3 + 1];
        float v2 = g[(size_t)src * 12 + p * 3 + 2] - tv[(size_t)dst * 3 + 2];
        const float* R = Rm + (size_t)dst * 9;
        float r0 = R[0] * v0 + R[3] * v1 + R[6] * v2;   // R^T * v
        float r1 = R[1] * v0 + R[4] * v1 + R[7] * v2;
        float r2 = R[2] * v0 + R[5] * v1 + R[8] * v2;
        relT[r * 16 + p * 3 + 0] = r0;
        relT[r * 16 + p * 3 + 1] = r1;
        relT[r * 16 + p * 3 + 2] = r2;
        relT[r * 16 + 12 + p] = sqrtf(r0 * r0 + r1 * r1 + r2 * r2 + 1e-8f);
    }
    __syncthreads();
    for (int i = tid; i < 16 * MINP; i += 128) {
        int r = i / MINP, c = i % MINP;
        float v;
        if (c < 256) v = s[(size_t)sDst[r] * CS + c];
        else if (c < 512) v = s[(size_t)sSrc[r] * CS + (c - 256)];
        else if (c < 640) v = z[(size_t)(e0 + r) * CZ + (c - 512)];
        else if (c < 656) v = relT[r * 16 + (c - 640)];
        else v = 0.f;
        Ami[i] = f2bf(v);
    }
    // stage 1
    v8f acc[4];
#pragma unroll
    for (int j = 0; j < 4; ++j) acc[j] = v8f_zero();
    int ct = wv * 64;
    for (int k0 = 0; k0 < MINP; k0 += 32) {
        __syncthreads();
        for (int i = tid; i < 256 * 32; i += 128) {
            int r = i >> 5, c = i & 31;
            sWt[i] = w1[(size_t)r * MINP + k0 + c];
        }
        __syncthreads();
        v16bf af = ld_afrag(Ami + k0, MINP, lane);
#pragma unroll
        for (int j = 0; j < 4; ++j)
            acc[j] = wmma_bf16(af, ld_bfrag(sWt + (ct + j * 16) * 32, lane), acc[j]);
    }
    int rbase = ((lane >> 4) << 3), cb = ct + (lane & 15);
#pragma unroll
    for (int j = 0; j < 4; ++j)
#pragma unroll
        for (int v = 0; v < 8; ++v) {
            int row = rbase + v, col = cb + j * 16;
            H[row * 256 + col] = f2bf(fmaxf(acc[j][v] + b1[col], 0.f));
        }
    // stage 2
#pragma unroll
    for (int j = 0; j < 4; ++j) acc[j] = v8f_zero();
    for (int k0 = 0; k0 < 256; k0 += 32) {
        __syncthreads();
        for (int i = tid; i < 256 * 32; i += 128) {
            int r = i >> 5, c = i & 31;
            sWt[i] = w2[(size_t)r * 256 + k0 + c];
        }
        __syncthreads();
        v16bf af = ld_afrag(H + k0, 256, lane);
#pragma unroll
        for (int j = 0; j < 4; ++j)
            acc[j] = wmma_bf16(af, ld_bfrag(sWt + (ct + j * 16) * 32, lane), acc[j]);
    }
#pragma unroll
    for (int j = 0; j < 4; ++j)
#pragma unroll
        for (int v = 0; v < 8; ++v) {
            int row = rbase + v, col = cb + j * 16;
            float val = (acc[j][v] + b2[col]) * (1.f / (float)KK);
            atomicAdd(&agg[(size_t)sDst[row] * CS + col], val);
        }
}

// ---------------------------------------------------------------------------
// FUSED edge update: zu = [s[dst]|s[src]|z] @ ew^T + eb ; z = LN(z+zu)
// 32 edges per block.
// ---------------------------------------------------------------------------
__global__ void edge_upd_kernel(const float* __restrict__ s, float* __restrict__ z,
                                const int* __restrict__ eidx,
                                const __bf16* __restrict__ ww, const float* __restrict__ wb,
                                const float* __restrict__ zg, const float* __restrict__ zb) {
    __shared__ __attribute__((aligned(16))) char smem[49152];
    __shared__ int sSrc[32], sDst[32];
    __shared__ float lmean[32], lrstd[32];
    __bf16* Ac   = (__bf16*)smem;              // 32 x 640
    __bf16* sWt  = (__bf16*)(smem + 40960);    // 128 x 32
    float*  outF = (float*)smem;               // 32 x 128 (reuses Ac after GEMM)
    int tid = threadIdx.x, lane = tid & 31, wv = tid >> 5;
    int e0 = blockIdx.x * 32;
    if (tid < 32) { int e = e0 + tid; sDst[tid] = e / KK; sSrc[tid] = eidx[e]; }
    __syncthreads();
    for (int i = tid; i < 32 * CATE; i += 128) {
        int r = i / CATE, c = i % CATE;
        float v;
        if (c < 256) v = s[(size_t)sDst[r] * CS + c];
        else if (c < 512) v = s[(size_t)sSrc[r] * CS + (c - 256)];
        else v = z[(size_t)(e0 + r) * CZ + (c - 512)];
        Ac[i] = f2bf(v);
    }
    v8f acc[4];
#pragma unroll
    for (int j = 0; j < 4; ++j) acc[j] = v8f_zero();
    int rt = (wv >> 1) * 16, ct = (wv & 1) * 64;
    for (int k0 = 0; k0 < CATE; k0 += 32) {
        __syncthreads();
        for (int i = tid; i < 128 * 32; i += 128) {
            int r = i >> 5, c = i & 31;
            sWt[i] = ww[(size_t)r * CATE + k0 + c];
        }
        __syncthreads();
        v16bf af = ld_afrag(Ac + rt * CATE + k0, CATE, lane);
#pragma unroll
        for (int j = 0; j < 4; ++j)
            acc[j] = wmma_bf16(af, ld_bfrag(sWt + (ct + j * 16) * 32, lane), acc[j]);
    }
    __syncthreads();   // Ac is dead; outF reuses its space
    int rbase = rt + ((lane >> 4) << 3), cb = ct + (lane & 15);
#pragma unroll
    for (int j = 0; j < 4; ++j)
#pragma unroll
        for (int v = 0; v < 8; ++v) {
            int row = rbase + v, col = cb + j * 16;
            outF[row * 128 + col] = acc[j][v] + wb[col];
        }
    __syncthreads();
    if (tid < 32) {
        float m = 0.f;
        for (int c = 0; c < 128; ++c)
            m += z[(size_t)(e0 + tid) * CZ + c] + outF[tid * 128 + c];
        m *= (1.f / 128.f);
        float vv = 0.f;
        for (int c = 0; c < 128; ++c) {
            float d = z[(size_t)(e0 + tid) * CZ + c] + outF[tid * 128 + c] - m;
            vv += d * d;
        }
        lmean[tid] = m;
        lrstd[tid] = rsqrtf(vv * (1.f / 128.f) + 1e-5f);
    }
    __syncthreads();
    for (int i = tid; i < 32 * 128; i += 128) {
        int r = i >> 7, c = i & 127;
        float v = z[(size_t)(e0 + r) * CZ + c] + outF[i];
        z[(size_t)(e0 + r) * CZ + c] = (v - lmean[r]) * lrstd[r] * zg[c] + zb[c];
    }
}

// ---------------------------------------------------------------------------
// host orchestration
// ---------------------------------------------------------------------------
enum {
    IN_XCA = 0, IN_BB, IN_MASK, IN_RIG, IN_BATCH,
    P_N0W, P_N0B, P_N1W, P_N1B, P_N2W, P_N2B, P_NLG, P_NLB,
    P_E0W, P_E0B, P_E1W, P_E1B, P_E2W, P_E2B, P_ELG, P_ELB,
    P_QW, P_QB, P_M1W, P_M1B, P_M2W, P_M2B, P_NW, P_NB, P_SG, P_SB,
    P_EW, P_EB, P_ZG, P_ZB, P_MUW, P_MUB, P_LVW, P_LVB
};

extern "C" void kernel_launch(void* const* d_in, const int* in_sizes, int n_in,
                              void* d_out, int out_size, void* d_ws, size_t ws_size,
                              hipStream_t stream) {
    (void)in_sizes; (void)n_in; (void)out_size; (void)ws_size;

    const float* x_ca  = (const float*)d_in[IN_XCA];
    const float* bb    = (const float*)d_in[IN_BB];
    const unsigned char* msk = (const unsigned char*)d_in[IN_MASK];
    const float* rig   = (const float*)d_in[IN_RIG];
    const int*   batch = (const int*)d_in[IN_BATCH];
    const float* PF[40];
    for (int i = P_N0W; i <= P_LVB; ++i) PF[i] = (const float*)d_in[i];
    float* outp = (float*)d_out;

    // ---- workspace layout
    char* ws = (char*)d_ws;
    size_t off = 0;
    auto take = [&](size_t bytes) -> char* {
        char* r = ws + off;
        off += (bytes + 255) & ~(size_t)255;
        return r;
    };
    __bf16* wb_n0w = (__bf16*)take((size_t)512 * 32 * 2);
    __bf16* wb_n1w = (__bf16*)take((size_t)512 * 512 * 2);
    __bf16* wb_n2w = (__bf16*)take((size_t)256 * 512 * 2);
    __bf16* wb_e0w = (__bf16*)take((size_t)256 * ZINP * 2);
    __bf16* wb_e1w = (__bf16*)take((size_t)256 * 256 * 2);
    __bf16* wb_e2w = (__bf16*)take((size_t)128 * 256 * 2);
    __bf16* wb_m1w = (__bf16*)take((size_t)LL * 256 * MINP * 2);
    __bf16* wb_m2w = (__bf16*)take((size_t)LL * 256 * 256 * 2);
    __bf16* wb_nw  = (__bf16*)take((size_t)LL * 256 * 512 * 2);
    __bf16* wb_ew  = (__bf16*)take((size_t)LL * 128 * CATE * 2);
    __bf16* wb_muw = (__bf16*)take((size_t)256 * 256 * 2);
    __bf16* wb_lvw = (__bf16*)take((size_t)256 * 256 * 2);
    float* bb5  = (float*)take((size_t)NN * 15 * 4);
    float* nfea = (float*)take((size_t)NN * 6 * 4);
    int*   eidx = (int*)  take((size_t)EE * 4);
    float* Rm   = (float*)take((size_t)NN * 9 * 4);
    float* tvb  = (float*)take((size_t)NN * 3 * 4);
    float* gbuf = (float*)take((size_t)NN * 12 * 4);
    float* sbuf = (float*)take((size_t)NN * CS * 4);
    float* zbuf = (float*)take((size_t)EE * CZ * 4);
    float* aggb = (float*)take((size_t)NN * CS * 4);
    float* h1   = (float*)take((size_t)NN * 512 * 4);
    float* h2   = (float*)take((size_t)NN * 512 * 4);
    float* updb = (float*)take((size_t)NN * CS * 4);

    auto cvt = [&](const float* src, __bf16* dst, int rows, int kin, int kp) {
        int tot = rows * kp;
        cvt_w_kernel<<<dim3((tot + 255) / 256), dim3(256), 0, stream>>>(src, dst, rows, kin, kp);
    };
    // ---- weight conversion (f32 -> bf16, K zero-padded)
    cvt(PF[P_N0W], wb_n0w, 512, 6, 32);
    cvt(PF[P_N1W], wb_n1w, 512, 512, 512);
    cvt(PF[P_N2W], wb_n2w, 256, 512, 512);
    cvt(PF[P_E0W], wb_e0w, 256, ZIN, ZINP);
    cvt(PF[P_E1W], wb_e1w, 256, 256, 256);
    cvt(PF[P_E2W], wb_e2w, 128, 256, 256);
    cvt(PF[P_M1W], wb_m1w, LL * 256, MIN_, MINP);
    cvt(PF[P_M2W], wb_m2w, LL * 256, 256, 256);
    cvt(PF[P_NW],  wb_nw,  LL * 256, 512, 512);
    cvt(PF[P_EW],  wb_ew,  LL * 128, CATE, CATE);
    cvt(PF[P_MUW], wb_muw, 256, 256, 256);
    cvt(PF[P_LVW], wb_lvw, 256, 256, 256);

    // ---- geometry prep
    prep_kernel<<<dim3((NN + 127) / 128), dim3(128), 0, stream>>>(bb, bb5, nfea);
    knn_kernel<<<dim3(NN), dim3(128), 0, stream>>>(x_ca, msk, batch, eidx);
    rigid_kernel<<<dim3((NN + 127) / 128), dim3(128), 0, stream>>>(rig, Rm, tvb);

    auto gemm = [&](const float* A, int lda, int M, int Kd, const __bf16* W, int ldw,
                    const float* bias, float* C, int ldc, int Nc, int relu) {
        gemm_kernel<<<dim3((M + 31) / 32, Nc / 64), dim3(128), 0, stream>>>(
            A, lda, M, Kd, W, ldw, bias, C, ldc, Nc, relu);
    };

    // ---- node MLP: s = LN(mlp3(nfeat)) * mask
    gemm(nfea, 6, NN, 6, wb_n0w, 32, PF[P_N0B], h1, 512, 512, 1);
    gemm(h1, 512, NN, 512, wb_n1w, 512, PF[P_N1B], h2, 512, 512, 1);
    gemm(h2, 512, NN, 512, wb_n2w, 512, PF[P_N2B], updb, 256, 256, 0);
    ln_rows_kernel<<<dim3(NN), dim3(128), 0, stream>>>(
        updb, (const float*)nullptr, sbuf, PF[P_NLG], PF[P_NLB], msk, CS);

    // ---- edge MLP (fused features + 3-layer MLP + LN) -> z
    edge_mlp_kernel<<<dim3(EE / 32), dim3(128), 0, stream>>>(
        bb5, eidx, wb_e0w, PF[P_E0B], wb_e1w, PF[P_E1B], wb_e2w, PF[P_E2B],
        PF[P_ELG], PF[P_ELB], zbuf);

    // ---- L layers
    for (int l = 0; l < LL; ++l) {
        const float* qw_l  = PF[P_QW] + (size_t)l * 12 * CS;
        const float* qb_l  = PF[P_QB] + (size_t)l * 12;
        const __bf16* m1w_l = wb_m1w + (size_t)l * 256 * MINP;
        const float* m1b_l = PF[P_M1B] + (size_t)l * 256;
        const __bf16* m2w_l = wb_m2w + (size_t)l * 256 * 256;
        const float* m2b_l = PF[P_M2B] + (size_t)l * 256;
        const __bf16* nw_l  = wb_nw + (size_t)l * 256 * 512;
        const float* nb_l  = PF[P_NB] + (size_t)l * 256;
        const __bf16* ew_l  = wb_ew + (size_t)l * 128 * CATE;
        const float* eb_l  = PF[P_EB] + (size_t)l * 128;

        pts_kernel<<<dim3((NN * PP + 127) / 128), dim3(128), 0, stream>>>(
            sbuf, qw_l, qb_l, Rm, tvb, gbuf);
        fill_zero_kernel<<<dim3((NN * CS + 255) / 256), dim3(256), 0, stream>>>(aggb, NN * CS);
        msg_kernel<<<dim3(EE / 16), dim3(128), 0, stream>>>(
            sbuf, zbuf, eidx, gbuf, Rm, tvb, m1w_l, m1b_l, m2w_l, m2b_l, aggb);
        cat_kernel<<<dim3((NN * 512 + 255) / 256), dim3(256), 0, stream>>>(sbuf, aggb, h1);
        gemm(h1, 512, NN, 512, nw_l, 512, nb_l, updb, 256, 256, 0);
        ln_rows_kernel<<<dim3(NN), dim3(128), 0, stream>>>(
            sbuf, updb, sbuf, PF[P_SG] + (size_t)l * 256, PF[P_SB] + (size_t)l * 256, msk, CS);
        edge_upd_kernel<<<dim3(EE / 32), dim3(128), 0, stream>>>(
            sbuf, zbuf, eidx, ew_l, eb_l,
            PF[P_ZG] + (size_t)l * 128, PF[P_ZB] + (size_t)l * 128);
    }

    // ---- heads: mu | lv concatenated in d_out
    gemm(sbuf, 256, NN, 256, wb_muw, 256, PF[P_MUB], outp, 256, 256, 0);
    gemm(sbuf, 256, NN, 256, wb_lvw, 256, PF[P_LVB], outp + (size_t)NN * CS, 256, 256, 0);
}
